// MLA_69166153334953
// MI455X (gfx1250) — matrix operand order repaired and checked by reading, not compiled, not run on hardware
//
#include <hip/hip_runtime.h>
#include <math.h>

// ---------------------------------------------------------------------------
// MLA prefill for MI455X (gfx1250, wave32, WMMA).
// fp32 in/out, bf16 WMMA (16x16x32) with fp32 accumulate. GEMMs stream the
// ~750MB of fp32 weights once (-> ~32us HBM floor @ 23.3 TB/s); M-blocks on
// blockIdx.x so co-resident workgroups share weight tiles through the 192MB
// L2. global_prefetch_b8 hides next-tile latency behind current-tile WMMAs.
// ---------------------------------------------------------------------------

#define S_   1024
#define HID_ 7168
#define NH_  128
#define DN_  128   // d_nope
#define DR_  64    // d_rope
#define DV_  128
#define QL_  1536  // q_lora
#define KVL_ 512   // kv_lora
#define CD_  (QL_ + KVL_ + DR_)   // 2112, comp width
#define QD_  (NH_ * (DN_ + DR_))  // 24576
#define KVD_ (NH_ * (DN_ + DV_))  // 32768
#define OD_  (NH_ * DV_)          // 16384

typedef __attribute__((ext_vector_type(16))) __bf16 v16bf;
typedef __attribute__((ext_vector_type(8)))  float  v8f;

union Frag16 { v16bf v; unsigned u[8]; };

__device__ __forceinline__ unsigned short f2bf(float f) {
  union { float f; unsigned u; } x; x.f = f;
  unsigned r = x.u + 0x7FFFu + ((x.u >> 16) & 1u);  // round-to-nearest-even
  return (unsigned short)(r >> 16);
}

__device__ __forceinline__ v8f wmma_bf16(v16bf a, v16bf b, v8f c) {
  // D = A(16x32 bf16) * B(32x16 bf16) + C(16x16 f32)
  return __builtin_amdgcn_wmma_f32_16x16x32_bf16(false, a, false, b,
                                                 (short)0, c, false, false);
}

// A-fragment (16x32 bf16) K index for VGPR j, lane-half (ISA 7.12.2)
__device__ __forceinline__ int a_kidx(int j, int half) {
  return (j < 4) ? (2 * j + 8 * half) : (8 + 2 * j + 8 * half);
}
// B-fragment (32x16 bf16) K index for VGPR j, lane-half
__device__ __forceinline__ int b_kidx(int j, int half) {
  return (j < 4) ? (2 * j + 8 * half) : (8 + 2 * j + 8 * half);
}

// ---------------------------------------------------------------------------
// Generic GEMM: C[M,N] = A[M,K] * B[K,N], fp32 global, bf16 LDS staging.
// Block: 256 thr (8 waves), tile 128x128x32, wave grid 2(M)x4(N).
// blockIdx.x = M-block (fastest) so concurrently-dispatched blocks share the
// same weight (B) strip -> B tiles hit L2 instead of re-streaming HBM.
// Requires K % 32 == 0 (true for all call sites); M,N guarded.
// ---------------------------------------------------------------------------
__global__ __launch_bounds__(256)
void gemm_bf16(const float* __restrict__ A, const float* __restrict__ B,
               float* __restrict__ C, int M, int N, int K) {
  constexpr int BM = 128, BN = 128, BK = 32, LS = BK + 8;  // padded bf16 stride
  __shared__ __align__(16) unsigned short As[BM * LS];
  __shared__ __align__(16) unsigned short Bs[BN * LS];     // transposed: [n][k]

  const int tid  = threadIdx.x;
  const int lane = tid & 31;
  const int wave = tid >> 5;
  const int wm   = wave >> 2;   // 0..1
  const int wn   = wave & 3;    // 0..3
  const int half = lane >> 4;
  const int l16  = lane & 15;
  const int gm0  = blockIdx.x * BM;   // M on x: co-resident blocks share B
  const int gn0  = blockIdx.y * BN;

  const v8f vzero = {0.f,0.f,0.f,0.f,0.f,0.f,0.f,0.f};
  v8f acc[4][2];
#pragma unroll
  for (int mt = 0; mt < 4; ++mt)
#pragma unroll
    for (int nt = 0; nt < 2; ++nt) acc[mt][nt] = vzero;

  for (int k0 = 0; k0 < K; k0 += BK) {
    const bool pf_ok = (k0 + BK) < K;
    // stage A tile (128x32 f32 -> bf16), 1024 float4 / 256 thr
#pragma unroll
    for (int i = 0; i < 4; ++i) {
      int idx = tid + i * 256;
      int r   = idx >> 3;            // 8 float4 per row
      int c   = (idx & 7) * 4;
      float4 v = make_float4(0.f, 0.f, 0.f, 0.f);
      int gr = gm0 + r;
      if (gr < M) {
        v = *(const float4*)(A + (size_t)gr * K + (k0 + c));
        if (pf_ok)  // prefetch next k-tile line (global_prefetch_b8)
          __builtin_prefetch(A + (size_t)gr * K + (k0 + BK + c), 0, 3);
      }
      unsigned short* d = &As[r * LS + c];
      d[0] = f2bf(v.x); d[1] = f2bf(v.y); d[2] = f2bf(v.z); d[3] = f2bf(v.w);
    }
    // stage B tile (32x128), transposed scatter into Bs[n][k]
#pragma unroll
    for (int i = 0; i < 4; ++i) {
      int idx = tid + i * 256;
      int kr  = idx >> 5;            // 32 float4 per k-row
      int c   = (idx & 31) * 4;
      int gn  = gn0 + c;
      const float* bp = B + (size_t)(k0 + kr) * N + gn;
      float4 v = make_float4(0.f, 0.f, 0.f, 0.f);
      if (gn + 3 < N) {
        v = *(const float4*)bp;
        if (pf_ok)
          __builtin_prefetch(B + (size_t)(k0 + BK + kr) * N + gn, 0, 3);
      } else {
        if (gn + 0 < N) v.x = bp[0];
        if (gn + 1 < N) v.y = bp[1];
        if (gn + 2 < N) v.z = bp[2];
      }
      Bs[(c + 0) * LS + kr] = f2bf(v.x);
      Bs[(c + 1) * LS + kr] = f2bf(v.y);
      Bs[(c + 2) * LS + kr] = f2bf(v.z);
      Bs[(c + 3) * LS + kr] = f2bf(v.w);
    }
    __syncthreads();

    Frag16 af[4];
#pragma unroll
    for (int mt = 0; mt < 4; ++mt) {
      int row = wm * 64 + mt * 16 + l16;
#pragma unroll
      for (int j = 0; j < 8; ++j)
        af[mt].u[j] = *(const unsigned*)&As[row * LS + a_kidx(j, half)];
    }
#pragma unroll
    for (int nt = 0; nt < 2; ++nt) {
      Frag16 bfr;
      int col = wn * 32 + nt * 16 + l16;
#pragma unroll
      for (int j = 0; j < 8; ++j)
        bfr.u[j] = *(const unsigned*)&Bs[col * LS + b_kidx(j, half)];
#pragma unroll
      for (int mt = 0; mt < 4; ++mt)
        acc[mt][nt] = wmma_bf16(af[mt].v, bfr.v, acc[mt][nt]);
    }
    __syncthreads();
  }

  // store C (C-layout: VGPR r -> rows r / r+8, lane -> N)
#pragma unroll
  for (int mt = 0; mt < 4; ++mt)
#pragma unroll
    for (int nt = 0; nt < 2; ++nt)
#pragma unroll
      for (int r = 0; r < 8; ++r) {
        int m = gm0 + wm * 64 + mt * 16 + r + 8 * half;
        int n = gn0 + wn * 32 + nt * 16 + l16;
        if (m < M && n < N) C[(size_t)m * N + n] = acc[mt][nt][r];
      }
}

// ---------------------------------------------------------------------------
// RMSNorm: one block per row. dst[s, 0:len] = x * rsqrt(mean(x^2)+eps) * gamma
// ---------------------------------------------------------------------------
__global__ __launch_bounds__(256)
void rmsnorm_kernel(const float* __restrict__ src, int stride, int len,
                    const float* __restrict__ gamma, float* __restrict__ dst) {
  __shared__ float red[8];
  const int s = blockIdx.x;
  const float* x = src + (size_t)s * stride;
  float* y = dst + (size_t)s * len;
  float ss = 0.f;
  for (int i = threadIdx.x; i < len; i += 256) { float v = x[i]; ss += v * v; }
#pragma unroll
  for (int o = 16; o > 0; o >>= 1) ss += __shfl_xor(ss, o, 32);
  if ((threadIdx.x & 31) == 0) red[threadIdx.x >> 5] = ss;
  __syncthreads();
  float tot = 0.f;
#pragma unroll
  for (int i = 0; i < 8; ++i) tot += red[i];
  float rs = rsqrtf(tot / (float)len + 1e-6f);
  for (int i = threadIdx.x; i < len; i += 256) y[i] = x[i] * rs * gamma[i];
}

// ---------------------------------------------------------------------------
// RoPE for k_pe: comp[s, 2048:2112] -> kpe[s, 0:64]. 32 threads handle pairs.
// ---------------------------------------------------------------------------
__global__ void rope_kpe_kernel(const float* __restrict__ comp,
                                float* __restrict__ kpe) {
  const int s = blockIdx.x, t = threadIdx.x;   // t in [0,32)
  const float* x = comp + (size_t)s * CD_ + (QL_ + KVL_);
  float inv = __powf(10000.f, -(float)t * (1.f / 32.f));
  float a = (float)s * inv;
  float si, c; __sincosf(a, &si, &c);
  float x1 = x[t], x2 = x[t + 32];
  kpe[(size_t)s * DR_ + t]      = x1 * c - x2 * si;
  kpe[(size_t)s * DR_ + t + 32] = x2 * c + x1 * si;
}

// RoPE for q_pe, in place on q[s, h, 128:192]
__global__ void rope_qpe_kernel(float* __restrict__ q) {
  const int s = blockIdx.x, h = blockIdx.y, t = threadIdx.x;
  float* x = q + ((size_t)s * NH_ + h) * (DN_ + DR_) + DN_;
  float inv = __powf(10000.f, -(float)t * (1.f / 32.f));
  float a = (float)s * inv;
  float si, c; __sincosf(a, &si, &c);
  float x1 = x[t], x2 = x[t + 32];
  x[t]      = x1 * c - x2 * si;
  x[t + 32] = x2 * c + x1 * si;
}

// ---------------------------------------------------------------------------
// Flash attention: one (64-q-row block, head) per workgroup, 4 waves.
// Per wave: 16 q-rows. d=192 (nope+rope), dv=128, key blocks of 64, causal.
// 160 WMMAs per K/V staging round (vs 40 in the 2-wave variant) -> barrier
// and staging overhead amortized 4x. ~77KB LDS (within 320KB WGP budget).
// ---------------------------------------------------------------------------
__global__ __launch_bounds__(128)
void mla_attn_kernel(const float* __restrict__ qbuf,
                     const float* __restrict__ kvbuf,
                     const float* __restrict__ kpe,
                     float* __restrict__ attnb) {
  constexpr int QS = 192 + 8;  // bf16 stride, even
  constexpr int VS = 64 + 8;
  __shared__ __align__(16) unsigned short Qs[4][16 * QS];
  __shared__ __align__(16) unsigned short Ks[64 * QS];      // [key n][d]
  __shared__ __align__(16) unsigned short Vt[128 * VS];     // [d][key n]
  __shared__ __align__(16) unsigned short Ps[4][16 * VS];   // per-wave P strip

  const int tid  = threadIdx.x;
  const int lane = tid & 31;
  const int wave = tid >> 5;     // 0..3
  const int half = lane >> 4;
  const int l16  = lane & 15;
  const int h    = blockIdx.y;
  const int qb   = blockIdx.x;
  const int s0   = qb * 64;
  const float scale = 0.07216878364870323f;  // 1/sqrt(192)
  const float NEG = -1e30f;

  // stage Q (64 rows x 192) as bf16, vectorized; q_pe already roped in qbuf
  for (int i = tid; i < 64 * 48; i += 128) {
    int r = i / 48, d = (i % 48) * 4;
    float4 v = *(const float4*)(qbuf + ((size_t)(s0 + r) * NH_ + h) * 192 + d);
    unsigned short* dst = &Qs[r >> 4][(r & 15) * QS + d];
    dst[0] = f2bf(v.x); dst[1] = f2bf(v.y);
    dst[2] = f2bf(v.z); dst[3] = f2bf(v.w);
  }
  __syncthreads();

  // Q A-fragments held in registers for the whole kernel (6 k-chunks of 32)
  Frag16 qf[6];
#pragma unroll
  for (int kc = 0; kc < 6; ++kc)
#pragma unroll
    for (int j = 0; j < 8; ++j)
      qf[kc].u[j] =
          *(const unsigned*)&Qs[wave][l16 * QS + kc * 32 + a_kidx(j, half)];

  const v8f vzero = {0.f,0.f,0.f,0.f,0.f,0.f,0.f,0.f};
  v8f o[8];
#pragma unroll
  for (int dt = 0; dt < 8; ++dt) o[dt] = vzero;
  float m_run[8], l_run[8];
#pragma unroll
  for (int r = 0; r < 8; ++r) { m_run[r] = NEG; l_run[r] = 0.f; }

  const int nkb = qb + 1;  // causal: key blocks 0..qb (uniform per block)
  for (int kb = 0; kb < nkb; ++kb) {
    const int k0 = kb * 64;
    // stage K block (64 x 192): d<128 from kv nope half, d>=128 roped k_pe
    for (int i = tid; i < 64 * 48; i += 128) {
      int n = i / 48, d = (i % 48) * 4;
      int k = k0 + n;
      float4 v;
      if (d < 128)
        v = *(const float4*)(kvbuf + ((size_t)k * NH_ + h) * 256 + d);
      else
        v = *(const float4*)(kpe + (size_t)k * DR_ + (d - 128));
      unsigned short* dst = &Ks[n * QS + d];
      dst[0] = f2bf(v.x); dst[1] = f2bf(v.y);
      dst[2] = f2bf(v.z); dst[3] = f2bf(v.w);
    }
    // stage V transposed: Vt[d][n]
    for (int i = tid; i < 64 * 32; i += 128) {
      int n = i / 32, d = (i % 32) * 4;
      float4 v = *(const float4*)
          (kvbuf + ((size_t)(k0 + n) * NH_ + h) * 256 + 128 + d);
      Vt[(d + 0) * VS + n] = f2bf(v.x);
      Vt[(d + 1) * VS + n] = f2bf(v.y);
      Vt[(d + 2) * VS + n] = f2bf(v.z);
      Vt[(d + 3) * VS + n] = f2bf(v.w);
    }
    __syncthreads();

    // S strip (16 rows x 64 keys): four 16x16 tiles, K-loop over 192
    v8f sacc[4];
#pragma unroll
    for (int nt = 0; nt < 4; ++nt) {
      sacc[nt] = vzero;
#pragma unroll
      for (int kc = 0; kc < 6; ++kc) {
        Frag16 bfr;
#pragma unroll
        for (int j = 0; j < 8; ++j)
          bfr.u[j] = *(const unsigned*)
              &Ks[(nt * 16 + l16) * QS + kc * 32 + b_kidx(j, half)];
        sacc[nt] = wmma_bf16(qf[kc].v, bfr.v, sacc[nt]);
      }
    }

    // scale + causal mask + online softmax (row = r + 8*half per 16-lane grp)
#pragma unroll
    for (int r = 0; r < 8; ++r) {
      int qrow = s0 + wave * 16 + r + 8 * half;
      float mx = NEG;
#pragma unroll
      for (int nt = 0; nt < 4; ++nt) {
        int kcol = k0 + nt * 16 + l16;
        float v = sacc[nt][r] * scale;
        if (kcol > qrow) v = NEG;
        sacc[nt][r] = v;
        mx = fmaxf(mx, v);
      }
#pragma unroll
      for (int off = 1; off < 16; off <<= 1)
        mx = fmaxf(mx, __shfl_xor(mx, off, 32));
      float mnew = fmaxf(m_run[r], mx);
      float corr = __expf(m_run[r] - mnew);
      m_run[r] = mnew;
      float rs = 0.f;
#pragma unroll
      for (int nt = 0; nt < 4; ++nt) {
        float p = __expf(sacc[nt][r] - mnew);
        sacc[nt][r] = p;
        rs += p;
      }
#pragma unroll
      for (int off = 1; off < 16; off <<= 1)
        rs += __shfl_xor(rs, off, 32);
      l_run[r] = l_run[r] * corr + rs;
#pragma unroll
      for (int dt = 0; dt < 8; ++dt) o[dt][r] *= corr;
      // spill P (C-layout) into [m][n] LDS strip for A-fragment reload
#pragma unroll
      for (int nt = 0; nt < 4; ++nt)
        Ps[wave][(r + 8 * half) * VS + nt * 16 + l16] = f2bf(sacc[nt][r]);
    }
    __syncthreads();  // P visible; block stays lockstep (nkb uniform)

    // O(16x128) += P(16x64) x V(64x128): 2 k-chunks
    Frag16 pf[2];
#pragma unroll
    for (int kc = 0; kc < 2; ++kc)
#pragma unroll
      for (int j = 0; j < 8; ++j)
        pf[kc].u[j] = *(const unsigned*)
            &Ps[wave][l16 * VS + kc * 32 + a_kidx(j, half)];
#pragma unroll
    for (int dt = 0; dt < 8; ++dt) {
#pragma unroll
      for (int kc = 0; kc < 2; ++kc) {
        Frag16 vf;
#pragma unroll
        for (int j = 0; j < 8; ++j)
          vf.u[j] = *(const unsigned*)
              &Vt[(dt * 16 + l16) * VS + kc * 32 + b_kidx(j, half)];
        o[dt] = wmma_bf16(pf[kc].v, vf.v, o[dt]);
      }
    }
    __syncthreads();  // before next iter overwrites Ks/Vt
  }

  // normalize and write out: attnb[s, h*128 + d]
#pragma unroll
  for (int dt = 0; dt < 8; ++dt)
#pragma unroll
    for (int r = 0; r < 8; ++r) {
      int m = s0 + wave * 16 + r + 8 * half;
      attnb[(size_t)m * OD_ + h * DV_ + dt * 16 + l16] = o[dt][r] / l_run[r];
    }
}

// ---------------------------------------------------------------------------
// Launch: comp -> rmsnorms + k_pe rope -> q/kv up-proj -> q_pe rope ->
//         flash attention -> output projection.
// Workspace (fp32): comp 1024x2112 | q_cn 1024x1536 | kv_cn 1024x512 |
//   q 1024x24576 | kv 1024x32768 | kpe 1024x64 | attn 1024x16384  (~305 MB)
// ---------------------------------------------------------------------------
extern "C" void kernel_launch(void* const* d_in, const int* in_sizes, int n_in,
                              void* d_out, int out_size, void* d_ws,
                              size_t ws_size, hipStream_t stream) {
  (void)in_sizes; (void)n_in; (void)out_size; (void)ws_size;
  const float* hidden     = (const float*)d_in[0];
  const float* w_kv_a     = (const float*)d_in[1];
  const float* q_a_gamma  = (const float*)d_in[2];
  const float* kv_a_gamma = (const float*)d_in[3];
  const float* w_q_b      = (const float*)d_in[4];
  const float* w_kv_b     = (const float*)d_in[5];
  const float* w_o        = (const float*)d_in[6];
  float* out = (float*)d_out;

  float* ws    = (float*)d_ws;
  float* comp  = ws;                          // S x 2112
  float* q_cn  = comp  + (size_t)S_ * CD_;    // S x 1536
  float* kv_cn = q_cn  + (size_t)S_ * QL_;    // S x 512
  float* qbuf  = kv_cn + (size_t)S_ * KVL_;   // S x 24576
  float* kvbuf = qbuf  + (size_t)S_ * QD_;    // S x 32768
  float* kpe   = kvbuf + (size_t)S_ * KVD_;   // S x 64
  float* attnb = kpe   + (size_t)S_ * DR_;    // S x 16384

  dim3 blk(256);
  // 1. comp = hidden @ w_kv_a   (1024 x 7168 x 2112)
  gemm_bf16<<<dim3(S_ / 128, (CD_ + 127) / 128), blk, 0, stream>>>(
      hidden, w_kv_a, comp, S_, CD_, HID_);
  // 2. rmsnorms + k_pe rope
  rmsnorm_kernel<<<S_, blk, 0, stream>>>(comp, CD_, QL_, q_a_gamma, q_cn);
  rmsnorm_kernel<<<S_, blk, 0, stream>>>(comp + QL_, CD_, KVL_, kv_a_gamma,
                                         kv_cn);
  rope_kpe_kernel<<<S_, 32, 0, stream>>>(comp, kpe);
  // 3. up-projections
  gemm_bf16<<<dim3(S_ / 128, QD_ / 128), blk, 0, stream>>>(
      q_cn, w_q_b, qbuf, S_, QD_, QL_);
  gemm_bf16<<<dim3(S_ / 128, KVD_ / 128), blk, 0, stream>>>(
      kv_cn, w_kv_b, kvbuf, S_, KVD_, KVL_);
  rope_qpe_kernel<<<dim3(S_, NH_), 32, 0, stream>>>(qbuf);
  // 4. attention (16 q-blocks x 128 heads, 4 waves each)
  mla_attn_kernel<<<dim3(S_ / 64, NH_), 128, 0, stream>>>(qbuf, kvbuf, kpe,
                                                          attnb);
  // 5. output projection (1024 x 16384 x 7168) -> d_out
  gemm_bf16<<<dim3(S_ / 128, HID_ / 128), blk, 0, stream>>>(
      attnb, w_o, out, S_, HID_, OD_);
}
